// MegatronAttention_71502615544036
// MI455X (gfx1250) — matrix operand order repaired and checked by reading, not compiled, not run on hardware
//
#include <hip/hip_runtime.h>

// ---------------- types ----------------
typedef __bf16 bhalf;
typedef __attribute__((ext_vector_type(16))) __bf16 v16bf;
typedef __attribute__((ext_vector_type(8)))  __bf16 v8bf;
typedef __attribute__((ext_vector_type(8)))  float  v8f;

// ---------------- problem constants ----------------
#define B_   2
#define S_   2048
#define H_   2048          // hidden dim
#define NH_  16
#define D_   128
#define F_   6144          // 3*NH_*D_
#define BS_  (B_ * S_)     // 4096
#define BH_  (B_ * NH_)    // 32

static __device__ __forceinline__ bhalf f2bf(float f) { return (bhalf)f; }

static __device__ __forceinline__ v16bf ldfrag(const bhalf* lo, const bhalf* hi) {
  v8bf a = *(const v8bf*)lo;
  v8bf b = *(const v8bf*)hi;
  return __builtin_shufflevector(a, b, 0,1,2,3,4,5,6,7,8,9,10,11,12,13,14,15);
}

static __device__ __forceinline__ v8f wmma_bf16(v16bf a, v16bf b, v8f c) {
  // D = A(16x32 bf16) x B(32x16 bf16) + C(16x16 f32)
  return __builtin_amdgcn_wmma_f32_16x16x32_bf16(false, a, false, b, (short)0, c,
                                                 false, false);
}

// ---- CDNA5 async copy global->LDS (ASYNCcnt path, CDNA5 ISA) ----
static __device__ __forceinline__ void async_ld_b128(unsigned lds_off,
                                                     const void* gptr) {
  unsigned long long ga = (unsigned long long)(size_t)gptr;
  asm volatile("global_load_async_to_lds_b128 %0, %1, off"
               :
               : "v"(lds_off), "v"(ga)
               : "memory");
}
static __device__ __forceinline__ void wait_async_le4() {
  asm volatile("s_wait_asynccnt 0x4" ::: "memory");
}
static __device__ __forceinline__ void wait_async_le0() {
  asm volatile("s_wait_asynccnt 0x0" ::: "memory");
}

static __device__ __forceinline__ float redmax16(float v) {
  v = fmaxf(v, __shfl_xor(v, 1, 32));
  v = fmaxf(v, __shfl_xor(v, 2, 32));
  v = fmaxf(v, __shfl_xor(v, 4, 32));
  v = fmaxf(v, __shfl_xor(v, 8, 32));
  return v;
}
static __device__ __forceinline__ float redsum16(float v) {
  v += __shfl_xor(v, 1, 32);
  v += __shfl_xor(v, 2, 32);
  v += __shfl_xor(v, 4, 32);
  v += __shfl_xor(v, 8, 32);
  return v;
}

// ---------------- elementwise / layout kernels ----------------
__global__ void cvt_f32_bf16_kernel(const float* __restrict__ in,
                                    bhalf* __restrict__ out, int n) {
  int idx = (blockIdx.x * blockDim.x + threadIdx.x) * 4;
  if (idx + 3 < n) {
    float4 v = *(const float4*)(in + idx);
    out[idx + 0] = f2bf(v.x);
    out[idx + 1] = f2bf(v.y);
    out[idx + 2] = f2bf(v.z);
    out[idx + 3] = f2bf(v.w);
  }
}

// in: [K][N] f32 row-major  ->  out: [N][K] bf16 row-major
__global__ __launch_bounds__(256) void transpose_cvt_kernel(
    const float* __restrict__ in, bhalf* __restrict__ out, int K, int N) {
  __shared__ float tile[32][33];
  int kx = blockIdx.x * 32, nx = blockIdx.y * 32;
  int tx = threadIdx.x & 31, ty = threadIdx.x >> 5;  // ty in 0..7
#pragma unroll
  for (int i = 0; i < 4; ++i)
    tile[ty + i * 8][tx] = in[(size_t)(kx + ty + i * 8) * N + nx + tx];
  __syncthreads();
#pragma unroll
  for (int i = 0; i < 4; ++i)
    out[(size_t)(nx + ty + i * 8) * K + kx + tx] = f2bf(tile[tx][ty + i * 8]);
}

// qkv f32 [b][s][3*NH*D] -> Qb,Kb bf16 [b][h][s][d]
__global__ void qk_extract_kernel(const float* __restrict__ qkv,
                                  bhalf* __restrict__ Qb,
                                  bhalf* __restrict__ Kb) {
  int idx = blockIdx.x * blockDim.x + threadIdx.x;   // B*H*S*D threads
  int d = idx & 127;
  int s = (idx >> 7) & 2047;
  int h = (idx >> 18) & 15;
  int b = idx >> 22;
  size_t src = ((size_t)(b * S_ + s)) * F_ + h * D_ + d;
  Qb[idx] = f2bf(qkv[src]);              // chunk 0: Q
  Kb[idx] = f2bf(qkv[src + NH_ * D_]);   // chunk 1: K
}

__global__ void zero_f32_kernel(float* p, int n) {
  int i = blockIdx.x * blockDim.x + threadIdx.x;
  if (i < n) p[i] = 0.f;
}

// values[b][s][h*D+d] = v[b][h][s][d] * colsum[b][h][s]   (reference's q-sum einsum)
__global__ void values_kernel(const float* __restrict__ qkv,
                              const float* __restrict__ colsum,
                              bhalf* __restrict__ vals) {
  int idx = blockIdx.x * blockDim.x + threadIdx.x;   // B*S*NH*D threads
  int d = idx & 127;
  int h = (idx >> 7) & 15;
  int s = (idx >> 11) & 2047;
  int b = idx >> 22;
  size_t src = ((size_t)(b * S_ + s)) * F_ + 2 * NH_ * D_ + h * D_ + d;
  vals[idx] = f2bf(qkv[src] * colsum[((size_t)(b * NH_ + h)) * S_ + s]);
}

// ---------------- WMMA GEMM: C[M][N] f32 = A[M][K] bf16 x Bt[N][K]^T ----------------
// Double-buffered LDS, tiles staged with global_load_async_to_lds_b128 (ASYNCcnt).
__global__ __launch_bounds__(256) void gemm_bf16_kernel(
    const bhalf* __restrict__ A,   // [M][K]
    const bhalf* __restrict__ Bt,  // [N][K]  (B stored transposed)
    float* __restrict__ C,         // [M][N]
    int M, int N, int K) {
  __shared__ bhalf As[2][128][40];   // 128x32 tile, +8 pad, x2 buffers (20 KB)
  __shared__ bhalf Bs[2][128][40];   // 20 KB

  const int tid  = threadIdx.x;
  const int lane = tid & 31, wid = tid >> 5;
  const int lr = lane & 15, hf = lane >> 4;
  const int kbA = hf * 8;    // A frag: K = kbA..kbA+7, kbA+16..kbA+23
  const int kbB = hf * 8;    // B frag: K = kbB..kbB+7, kbB+16..kbB+23

  const int bm = blockIdx.y * 128;
  const int bn = blockIdx.x * 128;
  const int wm = (wid & 3) * 32;   // 4 waves along M
  const int wn = (wid >> 2) * 64;  // 2 waves along N

  const int lrow = tid >> 1;        // 0..127
  const int lcol = (tid & 1) * 16;  // 0 or 16

  const bhalf* pa0 = A  + (size_t)(bm + lrow) * K + lcol;
  const bhalf* pb0 = Bt + (size_t)(bn + lrow) * K + lcol;

  auto issue = [&](int buf, int k0) {
    // 4 async b128 copies per thread: 16 bf16 of A-row + 16 bf16 of B-row
    async_ld_b128((unsigned)(size_t)&As[buf][lrow][lcol],     pa0 + k0);
    async_ld_b128((unsigned)(size_t)&As[buf][lrow][lcol + 8], pa0 + k0 + 8);
    async_ld_b128((unsigned)(size_t)&Bs[buf][lrow][lcol],     pb0 + k0);
    async_ld_b128((unsigned)(size_t)&Bs[buf][lrow][lcol + 8], pb0 + k0 + 8);
  };

  v8f acc[2][4] = {};
  const int KT = K >> 5;   // K-tiles of 32

  issue(0, 0);
  for (int kt = 0; kt < KT; ++kt) {
    const int buf = kt & 1;
    if (kt + 1 < KT) {
      issue(buf ^ 1, (kt + 1) << 5);   // prefetch next tile into other buffer
      wait_async_le4();                // batch for `buf` done (in-order completion)
    } else {
      wait_async_le0();
    }
    __syncthreads();                   // all waves' data landed in LDS[buf]

    v16bf af[2], bfr[4];
#pragma unroll
    for (int i = 0; i < 2; ++i) {
      const bhalf* p = &As[buf][wm + i * 16 + lr][kbA];
      af[i] = ldfrag(p, p + 16);
    }
#pragma unroll
    for (int j = 0; j < 4; ++j) {
      const bhalf* p = &Bs[buf][wn + j * 16 + lr][kbB];
      bfr[j] = ldfrag(p, p + 16);
    }
#pragma unroll
    for (int i = 0; i < 2; ++i)
#pragma unroll
      for (int j = 0; j < 4; ++j)
        acc[i][j] = wmma_bf16(af[i], bfr[j], acc[i][j]);

    __syncthreads();  // everyone done reading LDS[buf] before it is re-filled
  }

  // C layout: VGPR r -> row (r + 8*half), col = lane%16
#pragma unroll
  for (int i = 0; i < 2; ++i) {
    const int mr0 = bm + wm + i * 16 + hf * 8;
#pragma unroll
    for (int j = 0; j < 4; ++j) {
      const int nc = bn + wn + j * 16 + lr;
#pragma unroll
      for (int r = 0; r < 8; ++r)
        C[(size_t)(mr0 + r) * N + nc] = acc[i][j][r];
    }
  }
}

// ---------------- attention: colsum[b,h,k] = sum_q softmax(QK^T/sqrt(d))[q,k] ----------------
__global__ __launch_bounds__(128) void attn_colsum_kernel(
    const bhalf* __restrict__ Qb,   // [BH][S][D]
    const bhalf* __restrict__ Kb,   // [BH][S][D]
    float* __restrict__ colsum) {   // [BH][S], pre-zeroed
  const int lane = threadIdx.x & 31;
  const int wid  = threadIdx.x >> 5;
  const int lr = lane & 15;
  const int hf = lane >> 4;
  const int kbA = hf * 8;
  const int kbB = hf * 8;

  const int qt = blockIdx.x & 31;       // 32 q-tiles of 64 rows
  const int bh = blockIdx.x >> 5;       // 0..31
  const int qrow = qt * 64 + wid * 16;  // this wave's 16 q rows

  const bhalf* Qbase = Qb + ((size_t)bh * S_ + qrow) * D_;
  const bhalf* Kbase = Kb + (size_t)bh * S_ * D_;

  // Q fragments for the 4 K-dim slices of D=128 (held for whole kernel)
  v16bf qf[4];
#pragma unroll
  for (int s = 0; s < 4; ++s) {
    const bhalf* p = Qbase + (size_t)lr * D_ + s * 32 + kbA;
    qf[s] = ldfrag(p, p + 16);
  }

  float m[8], l[8];
#pragma unroll
  for (int r = 0; r < 8; ++r) { m[r] = -3.0e38f; l[r] = 0.f; }

  const float scale = 0.08838834764831845f;   // 1/sqrt(128)

  // pass 1: online row max + row sum
  for (int ct = 0; ct < S_ / 16; ++ct) {
    if (ct + 1 < S_ / 16)   // global_prefetch_b8 of next K column-tile
      __builtin_prefetch(Kbase + (size_t)((ct + 1) * 16 + lr) * D_ + kbB, 0, 3);
    v8f c = {};
#pragma unroll
    for (int s = 0; s < 4; ++s) {
      const bhalf* p = Kbase + (size_t)(ct * 16 + lr) * D_ + s * 32 + kbB;
      v16bf kf = ldfrag(p, p + 16);
      c = wmma_bf16(qf[s], kf, c);
    }
#pragma unroll
    for (int r = 0; r < 8; ++r) {
      float v  = c[r] * scale;
      float tm = redmax16(v);
      float nm = fmaxf(m[r], tm);
      float es = redsum16(__expf(v - nm));
      l[r] = l[r] * __expf(m[r] - nm) + es;
      m[r] = nm;
    }
  }

  float invl[8];
#pragma unroll
  for (int r = 0; r < 8; ++r) invl[r] = 1.0f / l[r];

  // pass 2: recompute scores, accumulate per-column prob sums
  for (int ct = 0; ct < S_ / 16; ++ct) {
    if (ct + 1 < S_ / 16)
      __builtin_prefetch(Kbase + (size_t)((ct + 1) * 16 + lr) * D_ + kbB, 0, 3);
    v8f c = {};
#pragma unroll
    for (int s = 0; s < 4; ++s) {
      const bhalf* p = Kbase + (size_t)(ct * 16 + lr) * D_ + s * 32 + kbB;
      v16bf kf = ldfrag(p, p + 16);
      c = wmma_bf16(qf[s], kf, c);
    }
    float cs = 0.f;
#pragma unroll
    for (int r = 0; r < 8; ++r)
      cs += __expf(c[r] * scale - m[r]) * invl[r];
    cs += __shfl_xor(cs, 16, 32);           // combine the two 16-lane halves
    if (hf == 0)
      atomicAdd(&colsum[(size_t)bh * S_ + ct * 16 + lr], cs);
  }
}

// ---------------- launch ----------------
extern "C" void kernel_launch(void* const* d_in, const int* in_sizes, int n_in,
                              void* d_out, int out_size, void* d_ws, size_t ws_size,
                              hipStream_t stream) {
  const float* x     = (const float*)d_in[0];
  const float* w_qkv = (const float*)d_in[1];
  const float* w_o   = (const float*)d_in[2];
  float* out = (float*)d_out;

  char* ws = (char*)d_ws;
  size_t off = 0;
  auto alloc = [&](size_t bytes) -> void* {
    void* p = ws + off;
    off += (bytes + 255) & ~(size_t)255;
    return p;
  };
  bhalf* xb     = (bhalf*)alloc((size_t)BS_ * H_ * 2);        // 16 MB
  bhalf* wqkvT  = (bhalf*)alloc((size_t)F_ * H_ * 2);         // 24 MB
  bhalf* woT    = (bhalf*)alloc((size_t)H_ * H_ * 2);         //  8 MB
  float* qkv    = (float*)alloc((size_t)BS_ * F_ * 4);        // 96 MB
  bhalf* Qb     = (bhalf*)alloc((size_t)BH_ * S_ * D_ * 2);   // 16 MB
  bhalf* Kb     = (bhalf*)alloc((size_t)BH_ * S_ * D_ * 2);   // 16 MB
  float* colsum = (float*)alloc((size_t)BH_ * S_ * 4);        // 256 KB
  bhalf* vals   = (bhalf*)alloc((size_t)BS_ * H_ * 2);        // 16 MB
  (void)ws_size; (void)in_sizes; (void)n_in; (void)out_size;

  // 1) x -> bf16
  cvt_f32_bf16_kernel<<<(BS_ * H_) / 1024, 256, 0, stream>>>(x, xb, BS_ * H_);
  // 2) weights -> bf16 transposed [N][K]
  transpose_cvt_kernel<<<dim3(H_ / 32, F_ / 32), 256, 0, stream>>>(w_qkv, wqkvT, H_, F_);
  transpose_cvt_kernel<<<dim3(H_ / 32, H_ / 32), 256, 0, stream>>>(w_o, woT, H_, H_);
  // 3) qkv = x @ w_qkv  (4096x6144x2048)
  gemm_bf16_kernel<<<dim3(F_ / 128, BS_ / 128), 256, 0, stream>>>(xb, wqkvT, qkv,
                                                                  BS_, F_, H_);
  // 4) extract Q,K -> bf16 [b,h,s,d]
  qk_extract_kernel<<<(BH_ * S_ * D_) / 256, 256, 0, stream>>>(qkv, Qb, Kb);
  // 5) colsum = 0; then softmax column sums via WMMA
  zero_f32_kernel<<<(BH_ * S_) / 256, 256, 0, stream>>>(colsum, BH_ * S_);
  attn_colsum_kernel<<<BH_ * (S_ / 64), 128, 0, stream>>>(Qb, Kb, colsum);
  // 6) values = V * colsum  -> bf16
  values_kernel<<<(BS_ * H_) / 256, 256, 0, stream>>>(qkv, colsum, vals);
  // 7) out = values @ w_o  (4096x2048x2048)
  gemm_bf16_kernel<<<dim3(H_ / 128, BS_ / 128), 256, 0, stream>>>(vals, woT, out,
                                                                  BS_, H_, H_);
}
